// Decoder_12249246728308
// MI455X (gfx1250) — matrix operand order repaired and checked by reading, not compiled, not run on hardware
//
#include <hip/hip_runtime.h>

// ---------------------------------------------------------------------------
// CDNA5 (gfx1250) WMMA transformer decoder forward pass.
// GEMM-shaped work -> v_wmma_f32_16x16x32_f16, 32x32 output slab per wave
// (4 WMMAs / K-step, 2x fragment reuse). Row-major tiles use the gfx1250
// async global->LDS path (ASYNCcnt) when the builtins exist, else float4.
// ---------------------------------------------------------------------------

typedef __attribute__((ext_vector_type(16))) _Float16 v16h;
typedef __attribute__((ext_vector_type(8)))  float    v8f;

static constexpr int Dm  = 512;   // model dim
static constexpr int Hh  = 8;     // heads
static constexpr int DHd = 64;    // head dim
static constexpr int SQq = 512;   // query length
static constexpr int SKk = 1024;  // encoder length
static constexpr int Ff  = 2048;  // ffn dim
static constexpr int Bb  = 4;     // batch
static constexpr int Ll  = 4;     // layers

#define LDSW 36  // 32 fp32 + 4 pad; 144B row stride: 16B aligned, bank-clean

#if __has_builtin(__builtin_amdgcn_global_load_async_to_lds_b128) && \
    __has_builtin(__builtin_amdgcn_s_wait_asynccnt)
#define HAVE_ASYNC_LDS 1
// Exact parameter types per hipcc diagnostic: pointee is a 16B int vector,
// source in addrspace(1) (printed "__device__"), dest in addrspace(3).
typedef int b128_t __attribute__((vector_size(16)));
typedef __attribute__((address_space(1))) b128_t* gptr_b128;
typedef __attribute__((address_space(3))) b128_t* lptr_b128;
#else
#define HAVE_ASYNC_LDS 0
#endif

// Fill NROWS x 32 fp32 tile (row-major source, leading dim ld) into LDS.
template <int NROWS>
__device__ __forceinline__ void fill_tile(float* lds, const float* g,
                                          size_t ld) {
  const int tid = threadIdx.x;
#if HAVE_ASYNC_LDS
#pragma unroll
  for (int i = 0; i < NROWS * 8 / 256; ++i) {  // 16B chunks
    int c = i * 256 + tid;
    int r = c >> 3, col = (c & 7) << 2;
    __builtin_amdgcn_global_load_async_to_lds_b128(
        (gptr_b128)(g + (size_t)r * ld + col),
        (lptr_b128)(lds + r * LDSW + col), 0, 0);
  }
#else
#pragma unroll
  for (int i = 0; i < NROWS * 4 / 256; ++i) {  // 32B chunks
    int c = i * 256 + tid;
    int r = c >> 2, col = (c & 3) << 3;
    const float* gp = g + (size_t)r * ld + col;
    float4 x0 = *(const float4*)gp;
    float4 x1 = *(const float4*)(gp + 4);
    __builtin_prefetch(gp + 32, 0, 1);  // next K-block
    float* dp = lds + r * LDSW + col;
    *(float4*)dp = x0;
    *(float4*)(dp + 4) = x1;
  }
#endif
}

__device__ __forceinline__ void tile_wait_and_barrier() {
#if HAVE_ASYNC_LDS
  __builtin_amdgcn_s_wait_asynccnt(0);
#endif
  __syncthreads();
}

// A 16x32 f16 fragment per ISA 7.12.2: lanes 0-15 hold K 0..7 & 16..23,
// lanes 16-31 hold K 8..15 & 24..31 (converted fp32 LDS -> f16 regs).
__device__ __forceinline__ v16h frag_a(const float* As, int row, int laneHi) {
  const float* p = As + row * LDSW + laneHi * 8;
  float4 a0 = *(const float4*)p;
  float4 a1 = *(const float4*)(p + 4);
  float4 a2 = *(const float4*)(p + 16);
  float4 a3 = *(const float4*)(p + 20);
  v16h f = {(_Float16)a0.x, (_Float16)a0.y, (_Float16)a0.z, (_Float16)a0.w,
            (_Float16)a1.x, (_Float16)a1.y, (_Float16)a1.z, (_Float16)a1.w,
            (_Float16)a2.x, (_Float16)a2.y, (_Float16)a2.z, (_Float16)a2.w,
            (_Float16)a3.x, (_Float16)a3.y, (_Float16)a3.z, (_Float16)a3.w};
  return f;
}

// B 32x16 f16 fragment: lanes 0-15 hold K 0..7 & 16..23, lanes 16-31 hold K 8..15 & 24..31.
__device__ __forceinline__ v16h frag_b(const float* Bs, int row, int laneHi) {
  const float* p = Bs + row * LDSW + laneHi * 8;
  float4 a0 = *(const float4*)p;
  float4 a1 = *(const float4*)(p + 4);
  float4 a2 = *(const float4*)(p + 16);
  float4 a3 = *(const float4*)(p + 20);
  v16h f = {(_Float16)a0.x, (_Float16)a0.y, (_Float16)a0.z, (_Float16)a0.w,
            (_Float16)a1.x, (_Float16)a1.y, (_Float16)a1.z, (_Float16)a1.w,
            (_Float16)a2.x, (_Float16)a2.y, (_Float16)a2.z, (_Float16)a2.w,
            (_Float16)a3.x, (_Float16)a3.y, (_Float16)a3.z, (_Float16)a3.w};
  return f;
}

__device__ __forceinline__ v8f wmma_f16(v16h a, v16h b, v8f c) {
  return __builtin_amdgcn_wmma_f32_16x16x32_f16(false, a, false, b, (short)0, c,
                                                false, false);
}

// ---------------------------------------------------------------------------
// C[M,N] = act(A[M,K] @ W[K,N] + bias[N]);  M%128==0, N%64==0, K%32==0
// grid = (N/64, M/128), block = 256 = 8 waves (4x2 of 32x32 slabs)
// ---------------------------------------------------------------------------
__global__ void __launch_bounds__(256)
wmma_gemm_bias_kernel(const float* __restrict__ A, const float* __restrict__ W,
                      const float* __restrict__ bias, float* __restrict__ C,
                      int M, int N, int K, int act) {
  __shared__ __align__(16) float As[128 * LDSW];
  __shared__ __align__(16) float Bs[64 * LDSW];

  const int tid = threadIdx.x;
  const int lane = tid & 31, wave = tid >> 5;
  const int lane16 = lane & 15, laneHi = lane >> 4;
  const int wm = wave & 3, wn = wave >> 2;
  const int m0 = blockIdx.y * 128, n0 = blockIdx.x * 64;
  const int bn = tid & 63, bk0 = (tid >> 6) << 3;

  v8f acc00 = {}, acc01 = {}, acc10 = {}, acc11 = {};

  for (int k0 = 0; k0 < K; k0 += 32) {
    fill_tile<128>(As, A + (size_t)m0 * K + k0, (size_t)K);
    {  // W[k][n] -> Bs[n][k] transpose gather (coalesced along n)
      const float* wp = W + (size_t)(k0 + bk0) * N + n0 + bn;
      float4 h0, h1;
      h0.x = wp[0];             h0.y = wp[(size_t)N];
      h0.z = wp[(size_t)2 * N]; h0.w = wp[(size_t)3 * N];
      h1.x = wp[(size_t)4 * N]; h1.y = wp[(size_t)5 * N];
      h1.z = wp[(size_t)6 * N]; h1.w = wp[(size_t)7 * N];
      float* dp = &Bs[bn * LDSW + bk0];
      *(float4*)dp = h0;
      *(float4*)(dp + 4) = h1;
    }
    tile_wait_and_barrier();

    v16h a0 = frag_a(As, wm * 32 + lane16, laneHi);
    v16h a1 = frag_a(As, wm * 32 + 16 + lane16, laneHi);
    v16h b0 = frag_b(Bs, wn * 32 + lane16, laneHi);
    v16h b1 = frag_b(Bs, wn * 32 + 16 + lane16, laneHi);
    acc00 = wmma_f16(a0, b0, acc00);
    acc01 = wmma_f16(a0, b1, acc01);
    acc10 = wmma_f16(a1, b0, acc10);
    acc11 = wmma_f16(a1, b1, acc11);
    __syncthreads();
  }

#pragma unroll
  for (int r = 0; r < 8; ++r) {
    int mi0 = m0 + wm * 32 + r + 8 * laneHi, mi1 = mi0 + 16;
    int ni0 = n0 + wn * 32 + lane16, ni1 = ni0 + 16;
    float b0 = bias[ni0], b1 = bias[ni1];
    float v00 = acc00[r] + b0, v01 = acc01[r] + b1;
    float v10 = acc10[r] + b0, v11 = acc11[r] + b1;
    if (act == 1) {
      v00 = fmaxf(v00, 0.f); v01 = fmaxf(v01, 0.f);
      v10 = fmaxf(v10, 0.f); v11 = fmaxf(v11, 0.f);
    }
    C[(size_t)mi0 * N + ni0] = v00;
    C[(size_t)mi0 * N + ni1] = v01;
    C[(size_t)mi1 * N + ni0] = v10;
    C[(size_t)mi1 * N + ni1] = v11;
  }
}

// ---------------------------------------------------------------------------
// S[bh,q,k] = (Qh[q,:] . Kh[k,:]) * scale + mask * -1e9
// grid = (Sk/64, Sq/128, B*H); maskmode 0: lookahead (Sq,Sk); 1: pad (B,Sk)
// Both operands are row-major with contiguous head dim -> both async-capable.
// ---------------------------------------------------------------------------
__global__ void __launch_bounds__(256)
wmma_attn_scores_kernel(const float* __restrict__ Q, const float* __restrict__ Kt,
                        const float* __restrict__ mask, float* __restrict__ S,
                        int Sq, int Sk, float scale, int maskmode) {
  __shared__ __align__(16) float As[128 * LDSW];
  __shared__ __align__(16) float Bs[64 * LDSW];

  const int tid = threadIdx.x;
  const int lane = tid & 31;
  const int wave = tid >> 5;
  const int lane16 = lane & 15, laneHi = lane >> 4;
  const int wm = wave & 3, wn = wave >> 2;
  const int q0 = blockIdx.y * 128, n0 = blockIdx.x * 64;
  const int bh = blockIdx.z, b = bh >> 3, h = bh & 7;

  const float* Qh = Q + (size_t)b * Sq * Dm + h * DHd;
  const float* Kh = Kt + (size_t)b * Sk * Dm + h * DHd;

  v8f acc00 = {}, acc01 = {}, acc10 = {}, acc11 = {};

#pragma unroll
  for (int k0 = 0; k0 < DHd; k0 += 32) {
    fill_tile<128>(As, Qh + (size_t)q0 * Dm + k0, (size_t)Dm);
    fill_tile<64>(Bs, Kh + (size_t)n0 * Dm + k0, (size_t)Dm);
    tile_wait_and_barrier();
    v16h a0 = frag_a(As, wm * 32 + lane16, laneHi);
    v16h a1 = frag_a(As, wm * 32 + 16 + lane16, laneHi);
    v16h b0 = frag_b(Bs, wn * 32 + lane16, laneHi);
    v16h b1 = frag_b(Bs, wn * 32 + 16 + lane16, laneHi);
    acc00 = wmma_f16(a0, b0, acc00);
    acc01 = wmma_f16(a0, b1, acc01);
    acc10 = wmma_f16(a1, b0, acc10);
    acc11 = wmma_f16(a1, b1, acc11);
    __syncthreads();
  }

#pragma unroll
  for (int r = 0; r < 8; ++r) {
    int qi0 = q0 + wm * 32 + r + 8 * laneHi, qi1 = qi0 + 16;
    int ni0 = n0 + wn * 32 + lane16, ni1 = ni0 + 16;
    float m00, m01, m10, m11;
    if (maskmode == 0) {
      m00 = mask[(size_t)qi0 * Sk + ni0]; m01 = mask[(size_t)qi0 * Sk + ni1];
      m10 = mask[(size_t)qi1 * Sk + ni0]; m11 = mask[(size_t)qi1 * Sk + ni1];
    } else {
      m00 = m10 = mask[(size_t)b * Sk + ni0];
      m01 = m11 = mask[(size_t)b * Sk + ni1];
    }
    size_t r0 = ((size_t)bh * Sq + qi0) * Sk, r1 = ((size_t)bh * Sq + qi1) * Sk;
    S[r0 + ni0] = acc00[r] * scale + m00 * -1e9f;
    S[r0 + ni1] = acc01[r] * scale + m01 * -1e9f;
    S[r1 + ni0] = acc10[r] * scale + m10 * -1e9f;
    S[r1 + ni1] = acc11[r] * scale + m11 * -1e9f;
  }
}

// ---------------------------------------------------------------------------
// O[bh,q,:] = P[bh,q,:] @ Vh[bh,:,:]   (N = DHd = 64)
// grid = (1, Sq/128, B*H). Also used for the pointer-gen context einsum.
// ---------------------------------------------------------------------------
__global__ void __launch_bounds__(256)
wmma_attn_pv_kernel(const float* __restrict__ P, const float* __restrict__ V,
                    float* __restrict__ O, int Sq, int Sk) {
  __shared__ __align__(16) float As[128 * LDSW];
  __shared__ __align__(16) float Bs[64 * LDSW];

  const int tid = threadIdx.x;
  const int lane = tid & 31, wave = tid >> 5;
  const int lane16 = lane & 15, laneHi = lane >> 4;
  const int wm = wave & 3, wn = wave >> 2;
  const int q0 = blockIdx.y * 128;
  const int bh = blockIdx.z, b = bh >> 3, h = bh & 7;
  const int bn = tid & 63, bk0 = (tid >> 6) << 3;

  const float* Ph = P + (size_t)bh * Sq * Sk;
  const float* Vh = V + (size_t)b * Sk * Dm + h * DHd;
  float* Oh = O + (size_t)b * Sq * Dm + h * DHd;

  v8f acc00 = {}, acc01 = {}, acc10 = {}, acc11 = {};

  for (int k0 = 0; k0 < Sk; k0 += 32) {
    fill_tile<128>(As, Ph + (size_t)q0 * Sk + k0, (size_t)Sk);
    {  // V[k][d] -> Bs[d][k] transpose gather
      const float* vp = Vh + (size_t)(k0 + bk0) * Dm + bn;
      float4 h0, h1;
      h0.x = vp[0];              h0.y = vp[(size_t)Dm];
      h0.z = vp[(size_t)2 * Dm]; h0.w = vp[(size_t)3 * Dm];
      h1.x = vp[(size_t)4 * Dm]; h1.y = vp[(size_t)5 * Dm];
      h1.z = vp[(size_t)6 * Dm]; h1.w = vp[(size_t)7 * Dm];
      float* dp = &Bs[bn * LDSW + bk0];
      *(float4*)dp = h0;
      *(float4*)(dp + 4) = h1;
    }
    tile_wait_and_barrier();
    v16h a0 = frag_a(As, wm * 32 + lane16, laneHi);
    v16h a1 = frag_a(As, wm * 32 + 16 + lane16, laneHi);
    v16h b0 = frag_b(Bs, wn * 32 + lane16, laneHi);
    v16h b1 = frag_b(Bs, wn * 32 + 16 + lane16, laneHi);
    acc00 = wmma_f16(a0, b0, acc00);
    acc01 = wmma_f16(a0, b1, acc01);
    acc10 = wmma_f16(a1, b0, acc10);
    acc11 = wmma_f16(a1, b1, acc11);
    __syncthreads();
  }

#pragma unroll
  for (int r = 0; r < 8; ++r) {
    int qi0 = q0 + wm * 32 + r + 8 * laneHi, qi1 = qi0 + 16;
    int ni0 = wn * 32 + lane16, ni1 = ni0 + 16;
    Oh[(size_t)qi0 * Dm + ni0] = acc00[r];
    Oh[(size_t)qi0 * Dm + ni1] = acc01[r];
    Oh[(size_t)qi1 * Dm + ni0] = acc10[r];
    Oh[(size_t)qi1 * Dm + ni1] = acc11[r];
  }
}

// ---------------------------------------------------------------------------
// In-place row softmax. grid.x = rows, block = 256.
// ---------------------------------------------------------------------------
__global__ void __launch_bounds__(256)
softmax_rows_kernel(float* __restrict__ S, int len) {
  __shared__ float red[256];
  float* row = S + (size_t)blockIdx.x * len;
  const int tid = threadIdx.x;

  float mx = -3.0e38f;
  for (int i = tid; i < len; i += 256) mx = fmaxf(mx, row[i]);
  red[tid] = mx;
  __syncthreads();
  for (int s = 128; s > 0; s >>= 1) {
    if (tid < s) red[tid] = fmaxf(red[tid], red[tid + s]);
    __syncthreads();
  }
  mx = red[0];
  __syncthreads();

  float sum = 0.f;
  for (int i = tid; i < len; i += 256) {
    float e = expf(row[i] - mx);
    row[i] = e;
    sum += e;
  }
  red[tid] = sum;
  __syncthreads();
  for (int s = 128; s > 0; s >>= 1) {
    if (tid < s) red[tid] += red[tid + s];
    __syncthreads();
  }
  float inv = 1.f / red[0];
  for (int i = tid; i < len; i += 256) row[i] *= inv;
}

// ---------------------------------------------------------------------------
// out[r,:] = LayerNorm(a[r,:] + res[r,:]) * g + beta.  D = 512, block = 256.
// ---------------------------------------------------------------------------
__global__ void __launch_bounds__(256)
add_layernorm_kernel(const float* __restrict__ a, const float* __restrict__ res,
                     const float* __restrict__ g, const float* __restrict__ beta,
                     float* __restrict__ out) {
  __shared__ float red[256];
  const int tid = threadIdx.x;
  const size_t base = (size_t)blockIdx.x * Dm;

  float x0 = a[base + tid] + res[base + tid];
  float x1 = a[base + tid + 256] + res[base + tid + 256];

  red[tid] = x0 + x1;
  __syncthreads();
  for (int s = 128; s > 0; s >>= 1) {
    if (tid < s) red[tid] += red[tid + s];
    __syncthreads();
  }
  float mean = red[0] * (1.f / Dm);
  __syncthreads();

  float d0 = x0 - mean, d1 = x1 - mean;
  red[tid] = d0 * d0 + d1 * d1;
  __syncthreads();
  for (int s = 128; s > 0; s >>= 1) {
    if (tid < s) red[tid] += red[tid + s];
    __syncthreads();
  }
  float rstd = rsqrtf(red[0] * (1.f / Dm) + 1e-6f);

  out[base + tid] = g[tid] * d0 * rstd + beta[tid];
  out[base + tid + 256] = g[tid + 256] * d1 * rstd + beta[tid + 256];
}

// ---------------------------------------------------------------------------
// p[r] = sigmoid(((x.wx + bx) + (o.ws + bs) + (c.wh + bh)) * vw + vb)
// grid.x = B*SQ tokens, block = 256.
// ---------------------------------------------------------------------------
__global__ void __launch_bounds__(256)
pgen_kernel(const float* __restrict__ x, const float* __restrict__ o,
            const float* __restrict__ c, const float* __restrict__ wx,
            const float* __restrict__ bx, const float* __restrict__ wsv,
            const float* __restrict__ bs, const float* __restrict__ wh,
            const float* __restrict__ bh, const float* __restrict__ vw,
            const float* __restrict__ vb, float* __restrict__ p) {
  __shared__ float red[256];
  const int tid = threadIdx.x;
  const size_t base = (size_t)blockIdx.x * Dm;

  float acc = x[base + tid] * wx[tid] + x[base + tid + 256] * wx[tid + 256] +
              o[base + tid] * wsv[tid] + o[base + tid + 256] * wsv[tid + 256] +
              c[base + tid] * wh[tid] + c[base + tid + 256] * wh[tid + 256];
  red[tid] = acc;
  __syncthreads();
  for (int s = 128; s > 0; s >>= 1) {
    if (tid < s) red[tid] += red[tid + s];
    __syncthreads();
  }
  if (tid == 0) {
    float sv = red[0] + bx[0] + bs[0] + bh[0];
    float z = sv * vw[0] + vb[0];
    p[blockIdx.x] = 1.f / (1.f + expf(-z));
  }
}

// ---------------------------------------------------------------------------
// Host orchestration
// ---------------------------------------------------------------------------
extern "C" void kernel_launch(void* const* d_in, const int* in_sizes, int n_in,
                              void* d_out, int out_size, void* d_ws,
                              size_t ws_size, hipStream_t stream) {
  const float* x   = (const float*)d_in[0];
  const float* enc = (const float*)d_in[1];
  const float* lam = (const float*)d_in[2];
  const float* pad = (const float*)d_in[3];
  const float* sa_wq = (const float*)d_in[4];
  const float* sa_bq = (const float*)d_in[5];
  const float* sa_wk = (const float*)d_in[6];
  const float* sa_bk = (const float*)d_in[7];
  const float* sa_wv = (const float*)d_in[8];
  const float* sa_bv = (const float*)d_in[9];
  const float* sa_wo = (const float*)d_in[10];
  const float* sa_bo = (const float*)d_in[11];
  const float* ca_wq = (const float*)d_in[12];
  const float* ca_bq = (const float*)d_in[13];
  const float* ca_wk = (const float*)d_in[14];
  const float* ca_bk = (const float*)d_in[15];
  const float* ca_wv = (const float*)d_in[16];
  const float* ca_bv = (const float*)d_in[17];
  const float* ca_wo = (const float*)d_in[18];
  const float* ca_bo = (const float*)d_in[19];
  const float* ffn_w1 = (const float*)d_in[20];
  const float* ffn_b1 = (const float*)d_in[21];
  const float* ffn_w2 = (const float*)d_in[22];
  const float* ffn_b2 = (const float*)d_in[23];
  const float* ln_g = (const float*)d_in[24];
  const float* ln_b = (const float*)d_in[25];
  const float* wx_w = (const float*)d_in[26];
  const float* wx_b = (const float*)d_in[27];
  const float* wsw  = (const float*)d_in[28];
  const float* wsb  = (const float*)d_in[29];
  const float* wh_w = (const float*)d_in[30];
  const float* wh_b = (const float*)d_in[31];
  const float* v_w  = (const float*)d_in[32];
  const float* v_b  = (const float*)d_in[33];

  const int Mq = Bb * SQq;  // 2048 decoder tokens
  const int Mk = Bb * SKk;  // 4096 encoder tokens
  const int BH = Bb * Hh;   // 32 (batch, head) pairs

  float* ws = (float*)d_ws;
  size_t off = 0;
  auto alloc = [&](size_t n) { float* p = ws + off; off += n; return p; };
  float* qb  = alloc((size_t)Mq * Dm);        // Q projections
  float* kb  = alloc((size_t)Mk * Dm);        // K projections (max: cross)
  float* vb  = alloc((size_t)Mk * Dm);        // V projections (max: cross)
  float* sS  = alloc((size_t)BH * SQq * SKk); // scores/probs (max: cross)
  float* att = alloc((size_t)Mq * Dm);        // attn out (pre out-proj)
  float* aa  = alloc((size_t)Mq * Dm);        // out-proj / ffn2 result
  float* o1  = alloc((size_t)Mq * Dm);
  float* o2  = alloc((size_t)Mq * Dm);
  float* hh  = alloc((size_t)Mq * Ff);        // ffn hidden
  float* outb = alloc((size_t)Mq * Dm);       // layer output (layers 0..2)
  float* ctx = alloc((size_t)Mq * Dm);        // pointer-gen context
  (void)ws_size; (void)n_in; (void)in_sizes; (void)out_size;

  float* finalOut = (float*)d_out;            // (B,SQ,D)
  float* pgens = finalOut + (size_t)Mq * Dm;  // (B,SQ,1)

  const dim3 blk(256);
  const float scale = 0.125f;  // 1/sqrt(DH)
  const float* cur = x;

  for (int l = 0; l < Ll; ++l) {
    const size_t wDD = (size_t)l * Dm * Dm, bD = (size_t)l * Dm;

    // ---- self attention ----
    wmma_gemm_bias_kernel<<<dim3(Dm / 64, Mq / 128), blk, 0, stream>>>(
        cur, sa_wq + wDD, sa_bq + bD, qb, Mq, Dm, Dm, 0);
    wmma_gemm_bias_kernel<<<dim3(Dm / 64, Mq / 128), blk, 0, stream>>>(
        cur, sa_wk + wDD, sa_bk + bD, kb, Mq, Dm, Dm, 0);
    wmma_gemm_bias_kernel<<<dim3(Dm / 64, Mq / 128), blk, 0, stream>>>(
        cur, sa_wv + wDD, sa_bv + bD, vb, Mq, Dm, Dm, 0);
    wmma_attn_scores_kernel<<<dim3(SQq / 64, SQq / 128, BH), blk, 0, stream>>>(
        qb, kb, lam, sS, SQq, SQq, scale, 0);
    softmax_rows_kernel<<<dim3(BH * SQq), blk, 0, stream>>>(sS, SQq);
    wmma_attn_pv_kernel<<<dim3(1, SQq / 128, BH), blk, 0, stream>>>(
        sS, vb, att, SQq, SQq);
    wmma_gemm_bias_kernel<<<dim3(Dm / 64, Mq / 128), blk, 0, stream>>>(
        att, sa_wo + wDD, sa_bo + bD, aa, Mq, Dm, Dm, 0);
    add_layernorm_kernel<<<dim3(Mq), blk, 0, stream>>>(
        aa, cur, ln_g + (size_t)(l * 3 + 0) * Dm,
        ln_b + (size_t)(l * 3 + 0) * Dm, o1);

    // ---- cross attention ----
    wmma_gemm_bias_kernel<<<dim3(Dm / 64, Mq / 128), blk, 0, stream>>>(
        o1, ca_wq + wDD, ca_bq + bD, qb, Mq, Dm, Dm, 0);
    wmma_gemm_bias_kernel<<<dim3(Dm / 64, Mk / 128), blk, 0, stream>>>(
        enc, ca_wk + wDD, ca_bk + bD, kb, Mk, Dm, Dm, 0);
    wmma_gemm_bias_kernel<<<dim3(Dm / 64, Mk / 128), blk, 0, stream>>>(
        enc, ca_wv + wDD, ca_bv + bD, vb, Mk, Dm, Dm, 0);
    wmma_attn_scores_kernel<<<dim3(SKk / 64, SQq / 128, BH), blk, 0, stream>>>(
        qb, kb, pad, sS, SQq, SKk, scale, 1);
    softmax_rows_kernel<<<dim3(BH * SQq), blk, 0, stream>>>(sS, SKk);
    wmma_attn_pv_kernel<<<dim3(1, SQq / 128, BH), blk, 0, stream>>>(
        sS, vb, att, SQq, SKk);
    wmma_gemm_bias_kernel<<<dim3(Dm / 64, Mq / 128), blk, 0, stream>>>(
        att, ca_wo + wDD, ca_bo + bD, aa, Mq, Dm, Dm, 0);
    add_layernorm_kernel<<<dim3(Mq), blk, 0, stream>>>(
        aa, o1, ln_g + (size_t)(l * 3 + 1) * Dm,
        ln_b + (size_t)(l * 3 + 1) * Dm, o2);

    // ---- feed-forward ----
    wmma_gemm_bias_kernel<<<dim3(Ff / 64, Mq / 128), blk, 0, stream>>>(
        o2, ffn_w1 + (size_t)l * Dm * Ff, ffn_b1 + (size_t)l * Ff, hh,
        Mq, Ff, Dm, 1 /*relu*/);
    wmma_gemm_bias_kernel<<<dim3(Dm / 64, Mq / 128), blk, 0, stream>>>(
        hh, ffn_w2 + (size_t)l * Ff * Dm, ffn_b2 + bD, aa, Mq, Dm, Ff, 0);
    float* lout = (l == Ll - 1) ? finalOut : outb;
    add_layernorm_kernel<<<dim3(Mq), blk, 0, stream>>>(
        aa, o2, ln_g + (size_t)(l * 3 + 2) * Dm,
        ln_b + (size_t)(l * 3 + 2) * Dm, lout);
    cur = lout;
  }

  // ---- pointer generator ----
  // sS still holds the LAST layer's cross-attn probabilities (block2).
  wmma_attn_pv_kernel<<<dim3(1, SQq / 128, BH), blk, 0, stream>>>(
      sS, enc, ctx, SQq, SKk);
  pgen_kernel<<<dim3(Mq), blk, 0, stream>>>(
      x, finalOut, ctx, wx_w, wx_b, wsw, wsb, wh_w, wh_b, v_w, v_b, pgens);
}
